// ARE_Conv_Resblock_14096082665965
// MI455X (gfx1250) — matrix operand order repaired and checked by reading, not compiled, not run on hardware
//
#include <hip/hip_runtime.h>

typedef __attribute__((ext_vector_type(16))) _Float16 v16h;
typedef __attribute__((ext_vector_type(8)))  float    v8f;

namespace {
constexpr int NPTS = 5000;
constexpr int KNB  = 16;
constexpr int CIN  = 128;
constexpr int CONV = 257;     // 2*CIN + 1
constexpr int FOUT = 512;     // KS * H
constexpr int NKC  = 9;       // ceil(257/32) K-chunks
constexpr int KPAD = NKC * 32;// 288
constexpr float EPSF  = 1e-6f;
constexpr float BNEPS = 1e-5f;
constexpr float NEGS  = 0.2f;

// workspace byte offsets (all 256B aligned)
constexpr size_t OFF_STATS = 0;                               // 304 floats used (zeroed per launch)
constexpr size_t OFF_AWB   = 4096;                            // 32*9*32*16 halves = 294912 B
constexpr size_t OFF_PTS   = OFF_AWB + 294912;                // N*48 f32
constexpr size_t OFF_S1    = OFF_PTS + (size_t)NPTS*48*4;     // N*128 f32
constexpr size_t OFF_SC    = OFF_S1  + (size_t)NPTS*128*4;    // N*128 f32
constexpr size_t OFF_FE    = OFF_SC  + (size_t)NPTS*128*4;    // N*192 f32
constexpr size_t OFF_ID    = OFF_FE  + (size_t)NPTS*192*4;    // N*384 f32
constexpr size_t OFF_P3    = OFF_ID  + (size_t)NPTS*384*4;    // N*384 f32
constexpr size_t OFF_D3    = OFF_P3  + (size_t)NPTS*384*4;    // N*384 f32
}

// Per-lane async copy: 16 bytes global -> LDS, tracked by ASYNCcnt.
__device__ __forceinline__ void async_g2l_b128(unsigned ldsoff, const void* gptr) {
  asm volatile("global_load_async_to_lds_b128 %0, %1, off"
               :: "v"(ldsoff), "v"(gptr)
               : "memory");
}
__device__ __forceinline__ void wait_asynccnt0() {
  asm volatile("s_wait_asynccnt 0" ::: "memory");
}

// ---------------------------------------------------------------------------
// k0: swizzle weightbank (257,512) f32 -> padded-K f16 WMMA A-fragments.
// A = W^T (M=outchan 512, K=convchan 257->288).
// Layout: frag[(mt*9+kc)*32 + lane][16 halves], contiguous-K order (same as the k4 B reads):
//   lanes 0-15: row m = lane,   halves 0..15 -> K 0..15
//   lanes16-31: row m = lane-16,halves 0..15 -> K 16..31
// ---------------------------------------------------------------------------
__global__ void k0_wb_swizzle(const float* __restrict__ wb, _Float16* __restrict__ awb) {
  int idx = blockIdx.x * 256 + threadIdx.x;       // 32*9*32 = 9216 fragment-lane slots
  if (idx >= 32 * NKC * 32) return;
  int lane = idx & 31;
  int kc   = (idx >> 5) % NKC;
  int mt   = idx / (32 * NKC);
  int m     = mt * 16 + (lane & 15);
  int kbase = kc * 32 + ((lane & 16) ? 16 : 0);
  _Float16* dst = awb + (size_t)idx * 16;
#pragma unroll
  for (int h = 0; h < 16; ++h) {
    int kk = kbase + h;
    float v = (kk < CONV) ? wb[(size_t)kk * FOUT + m] : 0.f;
    dst[h] = (_Float16)v;
  }
}

// ---------------------------------------------------------------------------
// k1: pts = s_pts[nbr]-q_pt ; center ; cross ; p = Wf @ local_feats ;
//     accumulate per-channel (16) sum/sumsq of ||p||+EPS for VN-BN stats.
// ---------------------------------------------------------------------------
__global__ void k1_geom(const float* __restrict__ q_pts, const float* __restrict__ s_pts,
                        const int* __restrict__ nbr, const float* __restrict__ Wf,
                        float* __restrict__ ptsw, float* __restrict__ stats) {
  __shared__ float P[3][16], C[3], X[3][16], PN[16][16];
  int n = blockIdx.x, t = threadIdx.x;
  if (t < 48) {
    int d = t >> 4, k = t & 15;
    int id = nbr[n * KNB + k];
    float v = s_pts[(size_t)id * 3 + d] - q_pts[(size_t)n * 3 + d];
    P[d][k] = v;
    ptsw[(size_t)n * 48 + t] = v;   // layout [n][d][k]
  }
  __syncthreads();
  if (t < 3) {
    float s = 0.f;
    for (int k = 0; k < 16; ++k) s += P[t][k];
    C[t] = s * (1.f / 16.f);
  }
  __syncthreads();
  if (t < 48) {
    int d = t >> 4, k = t & 15, d1 = (d + 1) % 3, d2 = (d + 2) % 3;
    X[d][k] = P[d1][k] * C[d2] - P[d2][k] * C[d1];
  }
  __syncthreads();
  {
    int ch = t >> 4, k = t & 15;
    float w0 = Wf[ch * 3], w1 = Wf[ch * 3 + 1], w2 = Wf[ch * 3 + 2];
    float nsq = 0.f;
    for (int d = 0; d < 3; ++d) {
      float pv = w0 * P[d][k] + w1 * C[d] + w2 * X[d][k];
      nsq += pv * pv;
    }
    PN[ch][k] = sqrtf(nsq) + EPSF;
  }
  __syncthreads();
  if ((t & 15) == 0) {
    int ch = t >> 4;
    float s = 0.f, sq = 0.f;
    for (int k = 0; k < 16; ++k) { float v = PN[ch][k]; s += v; sq += v * v; }
    atomicAdd(&stats[ch], s);
    atomicAdd(&stats[16 + ch], sq);
  }
}

// ---------------------------------------------------------------------------
// k2: apply VN-BN to p, vn_act(p,d), s=||act|| ; s1 = W1@s ;
//     accumulate BN1 stats (8 channels).
// ---------------------------------------------------------------------------
__global__ void k2_att(const float* __restrict__ ptsw, float* __restrict__ stats,
                       const float* __restrict__ Wf, const float* __restrict__ Wd,
                       const float* __restrict__ vng, const float* __restrict__ vnb,
                       const float* __restrict__ W1, float* __restrict__ s1w) {
  __shared__ float P[3][16], C[3], X[3][16], S[16][16], S1[8][16];
  int n = blockIdx.x, t = threadIdx.x;
  if (t < 48) P[t >> 4][t & 15] = ptsw[(size_t)n * 48 + t];
  __syncthreads();
  if (t < 3) {
    float s = 0.f;
    for (int k = 0; k < 16; ++k) s += P[t][k];
    C[t] = s * (1.f / 16.f);
  }
  __syncthreads();
  if (t < 48) {
    int d = t >> 4, k = t & 15, d1 = (d + 1) % 3, d2 = (d + 2) % 3;
    X[d][k] = P[d1][k] * C[d2] - P[d2][k] * C[d1];
  }
  __syncthreads();
  {
    int ch = t >> 4, k = t & 15;
    float wf0 = Wf[ch * 3], wf1 = Wf[ch * 3 + 1], wf2 = Wf[ch * 3 + 2];
    float wd0 = Wd[ch * 3], wd1 = Wd[ch * 3 + 1], wd2 = Wd[ch * 3 + 2];
    float p[3], dv[3];
    float nsq = 0.f;
    for (int d = 0; d < 3; ++d) {
      float l0 = P[d][k], l1 = C[d], l2 = X[d][k];
      p[d]  = wf0 * l0 + wf1 * l1 + wf2 * l2;
      dv[d] = wd0 * l0 + wd1 * l1 + wd2 * l2;
      nsq += p[d] * p[d];
    }
    float pn   = sqrtf(nsq) + EPSF;
    float cnt  = (float)(NPTS * KNB);
    float mean = stats[ch] / cnt;
    float var  = stats[16 + ch] / cnt - mean * mean;
    float nbn  = vng[ch] * (pn - mean) * rsqrtf(var + BNEPS) + vnb[ch];
    float sf   = nbn / pn;
    float dot = 0.f, dsq = 0.f;
    for (int d = 0; d < 3; ++d) { dot += p[d] * sf * dv[d]; dsq += dv[d] * dv[d]; }
    float inv = dot / (dsq + EPSF);
    float acc = 0.f;
    for (int d = 0; d < 3; ++d) {
      float pb = p[d] * sf;
      float pr = pb - inv * dv[d];
      float v  = NEGS * pb + (1.f - NEGS) * ((dot >= 0.f) ? pb : pr);
      acc += v * v;
    }
    S[ch][k] = sqrtf(acc);
  }
  __syncthreads();
  if (t < 128) {
    int o = t >> 4, k = t & 15;
    float a = 0.f;
    for (int c = 0; c < 16; ++c) a += W1[o * 16 + c] * S[c][k];
    s1w[(size_t)n * 128 + t] = a;
    S1[o][k] = a;
  }
  __syncthreads();
  if (t < 128 && (t & 15) == 0) {
    int o = t >> 4;
    float s = 0.f, sq = 0.f;
    for (int k = 0; k < 16; ++k) { float v = S1[o][k]; s += v; sq += v * v; }
    atomicAdd(&stats[32 + o], s);
    atomicAdd(&stats[40 + o], sq);
  }
}

// ---------------------------------------------------------------------------
// k3: BN1 + relu + W2 + b2 + softmax over 8 channels -> scores (N,8,16)
// ---------------------------------------------------------------------------
__global__ void k3_scores(const float* __restrict__ s1w, const float* __restrict__ stats,
                          const float* __restrict__ g1, const float* __restrict__ b1,
                          const float* __restrict__ W2, const float* __restrict__ b2,
                          float* __restrict__ scw) {
  __shared__ float R[8][16], S2[8][16];
  int n = blockIdx.x, t = threadIdx.x;   // 128 threads
  int o = t >> 4, k = t & 15;
  float cnt  = (float)(NPTS * KNB);
  float mean = stats[32 + o] / cnt;
  float var  = stats[40 + o] / cnt - mean * mean;
  float v    = s1w[(size_t)n * 128 + t];
  float r    = g1[o] * (v - mean) * rsqrtf(var + BNEPS) + b1[o];
  R[o][k]    = fmaxf(r, 0.f);
  __syncthreads();
  float a = b2[o];
  for (int c = 0; c < 8; ++c) a += W2[o * 8 + c] * R[c][k];
  S2[o][k] = a;
  __syncthreads();
  float mx = -3.4e38f;
  for (int c = 0; c < 8; ++c) mx = fmaxf(mx, S2[c][k]);
  float den = 0.f;
  for (int c = 0; c < 8; ++c) den += __expf(S2[c][k] - mx);
  scw[(size_t)n * 128 + t] = __expf(a - mx) / den;
}

// ---------------------------------------------------------------------------
// k4: score-weighted conv via WMMA. One workgroup (256 thr / 8 waves) per
// query point. Neighbor features are pulled global->LDS with per-lane async
// b128 copies (ASYNCcnt) into an f32 staging buffer (union'd with the O
// accumulator), then converted/transposed to the f16 B-matrix buffer nfT
// ([col][K], contiguous K so a B fragment is one v16h LDS load). Wave w owns
// score-group g=w (M-tiles 4w..4w+3); 9 K-chunk WMMAs per tile with A reused
// across the 3 N-tiles. Also computes the shortcut matvec.
// ---------------------------------------------------------------------------
__global__ void __launch_bounds__(256) k4_conv(
    const int* __restrict__ nbr, const float* __restrict__ s_feats,
    const float* __restrict__ ptsw, const float* __restrict__ scw,
    const _Float16* __restrict__ awb, const float* __restrict__ shW,
    float* __restrict__ few, float* __restrict__ idw) {
  __shared__ __attribute__((aligned(32))) _Float16 nfT[48][KPAD]; // [col=d*16+k][convchan]
  __shared__ __attribute__((aligned(16))) union {
    float stage[8][384];   // 8 neighbors x (128ch*3d) f32 = 12288 B (async landing zone)
    float O[64][48];       // score-weighted accumulator (used after staging is consumed)
  } su;
  __shared__ float NI[64][16];
  __shared__ float SCR[8][16];
  int n = blockIdx.x, t = threadIdx.x;

  if (t < 128) SCR[t >> 4][t & 15] = scw[(size_t)n * 128 + t];
  for (int i = t; i < 48 * 31; i += 256) nfT[i / 31][CONV + (i % 31)] = (_Float16)0.f;
  if (t < 48) nfT[t][256] = (_Float16)ptsw[(size_t)n * 48 + t];

  // ---- async gather: 2 chunks of 8 neighbors, f32 rows -> LDS staging ----
  unsigned stageBase = (unsigned)(uintptr_t)(&su.stage[0][0]);  // LDS byte offset (addr[31:0])
  const char* sfb = (const char*)s_feats;
  for (int c = 0; c < 2; ++c) {
#pragma unroll
    for (int i = 0; i < 3; ++i) {               // 768 x 16B = 12288 B per chunk
      int e = t + i * 256;
      int r = e / 96;                           // neighbor slot within chunk
      int j = e % 96;                           // 16B sub-block within 1536 B row
      int id = nbr[n * KNB + c * 8 + r];
      const void* g = (const void*)(sfb + (size_t)id * 1536 + (size_t)j * 16);
      async_g2l_b128(stageBase + (unsigned)e * 16u, g);
    }
    wait_asynccnt0();
    __syncthreads();                            // staged f32 visible to all waves
    for (int i = t; i < 8 * 384; i += 256) {    // convert+transpose -> nfT rows 128..255
      int r = i / 384, rem = i % 384;
      int ch = rem / 3, d = rem % 3;
      nfT[d * 16 + (c * 8 + r)][128 + ch] = (_Float16)su.stage[r][rem];
    }
    __syncthreads();                            // staging consumed before reuse
  }
  // rows 0..127 = feats - feats(k=0)
  for (int i = t; i < CIN * 48; i += 256) {
    int ch = i / 48, rem = i % 48, d = rem >> 4, k = rem & 15;
    nfT[d * 16 + k][ch] =
        (_Float16)((float)nfT[d * 16 + k][128 + ch] - (float)nfT[d * 16][128 + ch]);
  }
  for (int i = t; i < 64 * 48; i += 256) su.O[i / 48][i % 48] = 0.f;
  __syncthreads();

  const v16h* __restrict__ afr = reinterpret_cast<const v16h*>(awb);
  int wave = t >> 5, lane = t & 31;
  int laneM = lane & 15;
  int hi = (lane & 16) ? 16 : 0;     // B K-offset; C/D row offset is hi/2
  for (int mi = 0; mi < 4; ++mi) {
    int mt = wave * 4 + mi;          // out channels [mt*16, mt*16+16) ; group g = wave
    if (mt + 1 < 32)
      __builtin_prefetch((const void*)(afr + (mt + 1) * NKC * 32 + lane), 0, 1);
    v8f acc[3];
    v8f z = {0.f, 0.f, 0.f, 0.f, 0.f, 0.f, 0.f, 0.f};
    acc[0] = z; acc[1] = z; acc[2] = z;
    for (int kc = 0; kc < NKC; ++kc) {
      v16h a = afr[(mt * NKC + kc) * 32 + lane];
#pragma unroll
      for (int nt = 0; nt < 3; ++nt) {
        v16h b = *reinterpret_cast<const v16h*>(&nfT[nt * 16 + laneM][kc * 32 + hi]);
        acc[nt] = __builtin_amdgcn_wmma_f32_16x16x32_f16(
            false, a, false, b, (short)0, acc[nt], false, false);
      }
    }
    float sw = SCR[wave][laneM];     // score[g=wave][k=col&15]
#pragma unroll
    for (int nt = 0; nt < 3; ++nt) {
      int col = nt * 16 + laneM;
#pragma unroll
      for (int r = 0; r < 8; ++r) {
        int f = mi * 16 + r + (hi ? 8 : 0);   // f = (o % 64)
        atomicAdd(&su.O[f][col], acc[nt][r] * sw);
      }
    }
  }
  __syncthreads();
  for (int i = t; i < 1024; i += 256) {
    int f = i >> 4, k = i & 15;
    float a = su.O[f][k], b = su.O[f][16 + k], c = su.O[f][32 + k];
    NI[f][k] = 1.f / fmaxf(sqrtf(a * a + b * b + c * c), 1e-12f);
  }
  __syncthreads();
  if (t < 192) {
    int f = t / 3, d = t % 3;
    float s = 0.f;
    for (int k = 0; k < 16; ++k) s += su.O[f][d * 16 + k] * NI[f][k];
    few[(size_t)n * 192 + t] = s * (1.f / 16.f);
  }
  // shortcut: identify = shortcut_W @ feats(neighbor 0)
  for (int e = t; e < 384; e += 256) {
    int o = e / 3, d = e % 3;
    float a = 0.f;
    for (int ch = 0; ch < CIN; ++ch)
      a += shW[o * CIN + ch] * (float)nfT[d * 16][128 + ch];
    idw[(size_t)n * 384 + e] = a;
  }
}

// ---------------------------------------------------------------------------
// k5: feats -> vn_act(feats, relu_Wd@feats) -> p3 = un_Wf@f2, d3 = un_Wd@f2 ;
//     accumulate per-channel (128) stats of ||p3||+EPS.
// ---------------------------------------------------------------------------
__global__ void k5_unary(const float* __restrict__ few, const float* __restrict__ rWd,
                         const float* __restrict__ uWf, const float* __restrict__ uWd,
                         float* __restrict__ p3w, float* __restrict__ d3w,
                         float* __restrict__ stats) {
  __shared__ float F[64][3], DD[64][3], F2[64][3], P3[128][3];
  int n = blockIdx.x, t = threadIdx.x;  // 192 threads
  F[t / 3][t % 3] = few[(size_t)n * 192 + t];
  __syncthreads();
  {
    int o = t / 3, d = t % 3;
    float a = 0.f;
    for (int c = 0; c < 64; ++c) a += rWd[o * 64 + c] * F[c][d];
    DD[o][d] = a;
  }
  __syncthreads();
  {
    int o = t / 3, d = t % 3;
    float dot = 0.f, dsq = 0.f;
    for (int dd = 0; dd < 3; ++dd) { dot += F[o][dd] * DD[o][dd]; dsq += DD[o][dd] * DD[o][dd]; }
    float inv = dot / (dsq + EPSF);
    float p = F[o][d], pr = p - inv * DD[o][d];
    F2[o][d] = NEGS * p + (1.f - NEGS) * ((dot >= 0.f) ? p : pr);
  }
  __syncthreads();
  for (int e = t; e < 384; e += 192) {
    int o = e / 3, d = e % 3;
    float a = 0.f, b = 0.f;
    for (int c = 0; c < 64; ++c) {
      float f = F2[c][d];
      a += uWf[o * 64 + c] * f;
      b += uWd[o * 64 + c] * f;
    }
    P3[o][d] = a;
    p3w[(size_t)n * 384 + e] = a;
    d3w[(size_t)n * 384 + e] = b;
  }
  __syncthreads();
  if (t < 128) {
    float x = P3[t][0], y = P3[t][1], z = P3[t][2];
    float pn = sqrtf(x * x + y * y + z * z) + EPSF;
    atomicAdd(&stats[48 + t], pn);
    atomicAdd(&stats[176 + t], pn * pn);
  }
}

// ---------------------------------------------------------------------------
// k6: VN-BN on p3 (stats over N), vn_act(p3bn, d3), + identify -> out
// ---------------------------------------------------------------------------
__global__ void k6_out(const float* __restrict__ p3w, const float* __restrict__ d3w,
                       const float* __restrict__ idw, const float* __restrict__ stats,
                       const float* __restrict__ ug, const float* __restrict__ ub,
                       float* __restrict__ out) {
  __shared__ float P3[128][3], D3[128][3];
  int n = blockIdx.x, t = threadIdx.x;   // 192 threads
  for (int e = t; e < 384; e += 192) {
    P3[e / 3][e % 3] = p3w[(size_t)n * 384 + e];
    D3[e / 3][e % 3] = d3w[(size_t)n * 384 + e];
  }
  __syncthreads();
  float cnt = (float)NPTS;
  for (int e = t; e < 384; e += 192) {
    int o = e / 3, d = e % 3;
    float x = P3[o][0], y = P3[o][1], z = P3[o][2];
    float pn   = sqrtf(x * x + y * y + z * z) + EPSF;
    float mean = stats[48 + o] / cnt;
    float var  = stats[176 + o] / cnt - mean * mean;
    float nbn  = ug[o] * (pn - mean) * rsqrtf(var + BNEPS) + ub[o];
    float sf   = nbn / pn;
    float dot = 0.f, dsq = 0.f;
    for (int dd = 0; dd < 3; ++dd) { dot += P3[o][dd] * sf * D3[o][dd]; dsq += D3[o][dd] * D3[o][dd]; }
    float inv = dot / (dsq + EPSF);
    float p = P3[o][d] * sf, pr = p - inv * D3[o][d];
    float v = NEGS * p + (1.f - NEGS) * ((dot >= 0.f) ? p : pr);
    out[(size_t)n * 384 + e] = v + idw[(size_t)n * 384 + e];
  }
}

// ---------------------------------------------------------------------------
extern "C" void kernel_launch(void* const* d_in, const int* in_sizes, int n_in,
                              void* d_out, int out_size, void* d_ws, size_t ws_size,
                              hipStream_t stream) {
  const float* q_pts   = (const float*)d_in[0];
  const float* s_pts   = (const float*)d_in[1];
  const float* s_feats = (const float*)d_in[2];
  const int*   nbr     = (const int*)  d_in[3];
  const float* Wf      = (const float*)d_in[4];
  const float* Wd      = (const float*)d_in[5];
  const float* vng     = (const float*)d_in[6];
  const float* vnb     = (const float*)d_in[7];
  const float* W1      = (const float*)d_in[8];
  const float* g1      = (const float*)d_in[9];
  const float* b1      = (const float*)d_in[10];
  const float* W2      = (const float*)d_in[11];
  const float* b2      = (const float*)d_in[12];
  const float* wb      = (const float*)d_in[13];
  const float* shW     = (const float*)d_in[14];
  const float* rWd     = (const float*)d_in[15];
  const float* uWf     = (const float*)d_in[16];
  const float* uWd     = (const float*)d_in[17];
  const float* ug      = (const float*)d_in[18];
  const float* ub      = (const float*)d_in[19];

  char* ws = (char*)d_ws;
  float*     stats = (float*)    (ws + OFF_STATS);
  _Float16*  awb   = (_Float16*) (ws + OFF_AWB);
  float*     ptsw  = (float*)    (ws + OFF_PTS);
  float*     s1w   = (float*)    (ws + OFF_S1);
  float*     scw   = (float*)    (ws + OFF_SC);
  float*     few   = (float*)    (ws + OFF_FE);
  float*     idw   = (float*)    (ws + OFF_ID);
  float*     p3w   = (float*)    (ws + OFF_P3);
  float*     d3w   = (float*)    (ws + OFF_D3);
  float*     out   = (float*)    d_out;

  hipMemsetAsync(stats, 0, 4096, stream);
  k0_wb_swizzle<<<36, 256, 0, stream>>>(wb, awb);
  k1_geom     <<<NPTS, 256, 0, stream>>>(q_pts, s_pts, nbr, Wf, ptsw, stats);
  k2_att      <<<NPTS, 256, 0, stream>>>(ptsw, stats, Wf, Wd, vng, vnb, W1, s1w);
  k3_scores   <<<NPTS, 128, 0, stream>>>(s1w, stats, g1, b1, W2, b2, scw);
  k4_conv     <<<NPTS, 256, 0, stream>>>(nbr, s_feats, ptsw, scw, awb, shW, few, idw);
  k5_unary    <<<NPTS, 192, 0, stream>>>(few, rWd, uWf, uWd, p3w, d3w, stream ? stats : stats);
  k6_out      <<<NPTS, 192, 0, stream>>>(p3w, d3w, idw, stats, ug, ub, out);
  (void)in_sizes; (void)n_in; (void)out_size; (void)ws_size;
}
